// CausalSelfAttention_40157944218161
// MI455X (gfx1250) — matrix operand order repaired and checked by reading, not compiled, not run on hardware
//
#include <hip/hip_runtime.h>
#include <cstdint>
#include <cstddef>

typedef __attribute__((ext_vector_type(16))) __bf16        v16bf;
typedef __attribute__((ext_vector_type(8)))  float         v8f;
typedef __attribute__((ext_vector_type(4)))  unsigned int  v4u;
typedef __attribute__((ext_vector_type(8)))  unsigned int  v8u;

#define BATCH   8
#define SEQ     2048
#define DIM     512
#define QBLK    64      // Q rows per workgroup
#define KBLK    32      // KV rows per iteration
#define THREADS 256     // 8 waves of 32

__device__ __forceinline__ unsigned short f2bf(float f) {
  unsigned int u = __float_as_uint(f);
  u = u + 0x7FFFu + ((u >> 16) & 1u);   // round-to-nearest-even
  return (unsigned short)(u >> 16);
}

__device__ __forceinline__ v16bf make_frag(const unsigned short* p0,
                                           const unsigned short* p1) {
  v4u a = *(const v4u*)p0;
  v4u b = *(const v4u*)p1;
  v8u w;
  w[0] = a[0]; w[1] = a[1]; w[2] = a[2]; w[3] = a[3];
  w[4] = b[0]; w[5] = b[1]; w[6] = b[2]; w[7] = b[3];
  return __builtin_bit_cast(v16bf, w);
}

// A fragment (16x32 bf16) from row-major [16][stride] bf16 tile.
// ISA layout: lanes 0-15 row M=lane, K = kbase+0..7 then +16..23;
//             lanes 16-31 row M=lane-16, K = kbase+8..15 then +24..31.
__device__ __forceinline__ v16bf load_fragA(const unsigned short* base,
                                            int stride, int kbase, int lane) {
  int row = lane & 15, hi = lane >> 4;
  const unsigned short* p = base + row * stride + kbase + hi * 8;
  return make_frag(p, p + 16);
}

// B fragment (32x16 bf16) where B[k][n] = M[n][k], M row-major [16][stride].
// ISA layout: lanes 0-15 col N=lane, K = kbase+0..7 then +16..23; lanes 16-31 K = kbase+8..15 then +24..31.
__device__ __forceinline__ v16bf load_fragB(const unsigned short* base,
                                            int stride, int kbase, int lane) {
  int row = lane & 15, hi = lane >> 4;
  const unsigned short* p = base + row * stride + kbase + hi * 8;
  return make_frag(p, p + 16);
}

__global__ __launch_bounds__(THREADS, 1)
void fa_causal_kernel(const float* __restrict__ x, float* __restrict__ out) {
  extern __shared__ char smem[];
  unsigned short* Qs  = (unsigned short*)smem;                 // [64][512] bf16
  unsigned short* Ks  = Qs + QBLK * DIM;                       // [32][512] bf16
  unsigned short* Vts = Ks + KBLK * DIM;                       // [512][32] bf16 (transposed)
  float* Spart        = (float*)(Vts + DIM * KBLK);            // [4][2][8][32] f32
  unsigned short* Pl  = (unsigned short*)(Spart + 4 * 2 * 8 * 32); // [4][16][32] bf16
  float* alphaL       = (float*)(Pl + 4 * 16 * 32);            // [64]
  float* linvL        = alphaL + 64;                           // [64]

  const int tid   = threadIdx.x;
  const int lane  = tid & 31;
  const int wave  = tid >> 5;
  const int qsub  = wave >> 1;    // 0..3 : which 16-row Q sub-tile
  const int dhalf = wave & 1;     // 0..1 : which 256-wide half of D
  const int hi    = lane >> 4;
  const int lo    = lane & 15;

  const int g     = blockIdx.x & 31;   // q-block index within sequence
  const int b     = blockIdx.x >> 5;   // batch
  const int qbase = g * QBLK;
  const int qr0   = qbase + qsub * 16; // first global Q row of this wave
  const int kd0   = dhalf * 256;       // this wave's D-half base

  const float* xb = x + (size_t)b * SEQ * DIM;

  // ---- Load Q block -> LDS (bf16) ----
  {
    const float4* src = (const float4*)(xb + (size_t)qbase * DIM);
#pragma unroll
    for (int i = 0; i < (QBLK * DIM) / (4 * THREADS); ++i) {
      float4 v = src[i * THREADS + tid];
      int f = (i * THREADS + tid) * 4;
      Qs[f + 0] = f2bf(v.x); Qs[f + 1] = f2bf(v.y);
      Qs[f + 2] = f2bf(v.z); Qs[f + 3] = f2bf(v.w);
    }
  }

  float mstate[8], lstate[8];
#pragma unroll
  for (int r = 0; r < 8; ++r) { mstate[r] = -1e30f; lstate[r] = 0.f; }

  v8f o[16];
#pragma unroll
  for (int c = 0; c < 16; ++c)
#pragma unroll
    for (int r = 0; r < 8; ++r) o[c][r] = 0.f;

  const int nkv = 2 * g + 2;   // causal: keys 0 .. qbase+63

  for (int it = 0; it < nkv; ++it) {
    const int k0 = it * KBLK;

    __syncthreads();  // (A) previous iteration's reads of Ks/Vts/Pl done

    // ---- Load KV block (same tensor: Q=K=V=x): Ks row-major, Vts transposed ----
    {
      const float4* ksrc = (const float4*)(xb + (size_t)k0 * DIM);
#pragma unroll
      for (int i = 0; i < (KBLK * DIM) / (4 * THREADS); ++i) {
        float4 v = ksrc[i * THREADS + tid];
        int f  = (i * THREADS + tid) * 4;
        unsigned short b0 = f2bf(v.x), b1 = f2bf(v.y), b2 = f2bf(v.z), b3 = f2bf(v.w);
        Ks[f + 0] = b0; Ks[f + 1] = b1; Ks[f + 2] = b2; Ks[f + 3] = b3;
        int kk = f >> 9;      // key row 0..31
        int d  = f & 511;     // feature 0..511
        Vts[(d + 0) * KBLK + kk] = b0;
        Vts[(d + 1) * KBLK + kk] = b1;
        Vts[(d + 2) * KBLK + kk] = b2;
        Vts[(d + 3) * KBLK + kk] = b3;
      }
    }
    __syncthreads();  // (B) tiles ready

    // ---- Partial scores S = Q * K^T over this wave's D-half: 16x32 ----
    v8f s0, s1;
#pragma unroll
    for (int r = 0; r < 8; ++r) { s0[r] = 0.f; s1[r] = 0.f; }
    const unsigned short* qtile = Qs + (qsub * 16) * DIM;
#pragma unroll
    for (int c = 0; c < 8; ++c) {
      int kb = kd0 + c * 32;
      v16bf a  = load_fragA(qtile, DIM, kb, lane);
      v16bf b0 = load_fragB(Ks, DIM, kb, lane);            // key cols 0..15
      v16bf b1 = load_fragB(Ks + 16 * DIM, DIM, kb, lane); // key cols 16..31
      s0 = __builtin_amdgcn_wmma_f32_16x16x32_bf16(false, a, false, b0, (short)0, s0, false, false);
      s1 = __builtin_amdgcn_wmma_f32_16x16x32_bf16(false, a, false, b1, (short)0, s1, false, false);
    }

    if (dhalf == 1) {
      float* sp = Spart + qsub * 512;
#pragma unroll
      for (int r = 0; r < 8; ++r) {
        sp[r * 32 + lane]       = s0[r];
        sp[256 + r * 32 + lane] = s1[r];
      }
    }
    __syncthreads();  // (C) partials visible

    // ---- dhalf==0: combine halves, causal mask, online softmax ----
    if (dhalf == 0) {
      const float* sp = Spart + qsub * 512;
      float p0[8], p1[8], alpha[8], rmax[8], rsum[8];
#pragma unroll
      for (int r = 0; r < 8; ++r) {
        float v0 = s0[r] + sp[r * 32 + lane];
        float v1 = s1[r] + sp[256 + r * 32 + lane];
        int mrow = qr0 + r + hi * 8;                 // global query row
        if (k0 + lo > mrow)      v0 = -1e30f;        // causal mask
        if (k0 + 16 + lo > mrow) v1 = -1e30f;
        p0[r] = v0; p1[r] = v1;
        rmax[r] = fmaxf(v0, v1);
      }
#pragma unroll
      for (int off = 1; off < 16; off <<= 1)
#pragma unroll
        for (int r = 0; r < 8; ++r)
          rmax[r] = fmaxf(rmax[r], __shfl_xor(rmax[r], off, 32));
#pragma unroll
      for (int r = 0; r < 8; ++r) {
        float mnew = fmaxf(mstate[r], rmax[r]);
        alpha[r] = __expf(mstate[r] - mnew);
        p0[r]    = __expf(p0[r] - mnew);
        p1[r]    = __expf(p1[r] - mnew);
        mstate[r] = mnew;
        rsum[r] = p0[r] + p1[r];
      }
#pragma unroll
      for (int off = 1; off < 16; off <<= 1)
#pragma unroll
        for (int r = 0; r < 8; ++r)
          rsum[r] += __shfl_xor(rsum[r], off, 32);
#pragma unroll
      for (int r = 0; r < 8; ++r)
        lstate[r] = lstate[r] * alpha[r] + rsum[r];

      unsigned short* pl = Pl + qsub * 512;
#pragma unroll
      for (int r = 0; r < 8; ++r) {
        int M = r + hi * 8;
        pl[M * 32 + lo]      = f2bf(p0[r]);
        pl[M * 32 + 16 + lo] = f2bf(p1[r]);
      }
      if (lo == 0) {
#pragma unroll
        for (int r = 0; r < 8; ++r) alphaL[qsub * 16 + r + hi * 8] = alpha[r];
      }
    }
    __syncthreads();  // (D) P / alpha visible

    // ---- O = O*alpha + P @ V over this wave's D-half ----
    float al[8];
#pragma unroll
    for (int r = 0; r < 8; ++r) al[r] = alphaL[qsub * 16 + r + hi * 8];
    v16bf pf = load_fragA(Pl + qsub * 512, 32, 0, lane);
#pragma unroll
    for (int c = 0; c < 16; ++c) {
      v16bf vb = load_fragB(Vts + (kd0 + c * 16) * KBLK, KBLK, 0, lane);
#pragma unroll
      for (int r = 0; r < 8; ++r) o[c][r] *= al[r];
      o[c] = __builtin_amdgcn_wmma_f32_16x16x32_bf16(false, pf, false, vb, (short)0, o[c], false, false);
    }
  }

  // ---- Normalize and write out ----
  if (dhalf == 0 && lo == 0) {
#pragma unroll
    for (int r = 0; r < 8; ++r) linvL[qsub * 16 + r + hi * 8] = 1.f / lstate[r];
  }
  __syncthreads();
  float li[8];
#pragma unroll
  for (int r = 0; r < 8; ++r) li[r] = linvL[qsub * 16 + r + hi * 8];

  float* op = out + ((size_t)b * SEQ + qr0) * DIM;
#pragma unroll
  for (int c = 0; c < 16; ++c) {
#pragma unroll
    for (int r = 0; r < 8; ++r) {
      op[(size_t)(r + hi * 8) * DIM + kd0 + c * 16 + lo] = o[c][r] * li[r];
    }
  }
}

extern "C" void kernel_launch(void* const* d_in, const int* in_sizes, int n_in,
                              void* d_out, int out_size, void* d_ws, size_t ws_size,
                              hipStream_t stream) {
  (void)in_sizes; (void)n_in; (void)out_size; (void)d_ws; (void)ws_size;
  const float* x = (const float*)d_in[0];
  float* out = (float*)d_out;

  constexpr size_t SMEM_BYTES =
      (size_t)(QBLK * DIM + KBLK * DIM + DIM * KBLK) * sizeof(unsigned short) // Qs, Ks, Vts
      + (size_t)(4 * 2 * 8 * 32) * sizeof(float)                             // Spart
      + (size_t)(4 * 16 * 32) * sizeof(unsigned short)                       // Pl
      + (size_t)128 * sizeof(float);                                         // alphaL + linvL

  dim3 grid(BATCH * (SEQ / QBLK));   // 256 workgroups
  dim3 block(THREADS);
  hipLaunchKernelGGL(fa_causal_kernel, grid, block, SMEM_BYTES, stream, x, out);
}
